// coverage_9569187135781
// MI455X (gfx1250) — matrix operand order repaired and checked by reading, not compiled, not run on hardware
//
#include <hip/hip_runtime.h>
#include <hip/hip_bf16.h>
#include <math.h>

// ---------------------------------------------------------------------------
// CDNA5 (gfx1250) attention+coverage seq2seq forward.
// All batchable matmuls -> WMMA GEMMs (f16 in / f32 acc). Since M==96 for
// every GEMM here, each block covers the FULL M dimension (96 = 6 x 16-row
// tiles): weights stream from HBM exactly once (196 MB for the final logits
// GEMM). K loop is software-pipelined: next tile loads in registers overlap
// the current tile's 6 v_wmma ops. Sequential 512-wide recurrences run in
// single-WGP kernels with float4 weight streaming.
// ---------------------------------------------------------------------------

typedef __attribute__((ext_vector_type(16))) _Float16 v16h;
typedef __attribute__((ext_vector_type(4)))  _Float16 v4h;
typedef __attribute__((ext_vector_type(8)))  float    v8f;

#define SLEN 96
#define EMB  256
#define HID  512
#define LDSW 40   // LDS row stride in halves (32 data + 8 pad); 80B rows, 16B-aligned

__device__ __forceinline__ float sigmoidf_(float x) {
  return 1.0f / (1.0f + __expf(-x));
}

__device__ __forceinline__ v4h cvt4(float4 v) {
  v4h h = {(_Float16)v.x, (_Float16)v.y, (_Float16)v.z, (_Float16)v.w};
  return h;
}

// ---------------------------------------------------------------------------
// WMMA GEMM:  C[96,N] = A[96,K] @ W[N,K]^T (+bias[n]) (+addMat[m,n]) (+addRow[n])
// Block = 128 threads (4 wave32 waves), grid = N/64. Block tile = 96 x 64;
// wave w owns the 16-wide N slice w and iterates 6 M-tiles -> 6 v_wmma per
// 32-deep K step. Pipelined: register double-buffer hides global latency
// behind the WMMAs. Requires N%64==0, K%32==0 (call sites: N in
// {512,1536,32000}, K in {256,512,1024,1536}).
// ---------------------------------------------------------------------------
__global__ __launch_bounds__(128) void wmma_gemm_kernel(
    const float* __restrict__ A, const float* __restrict__ W,
    const float* __restrict__ bias, const float* __restrict__ addMat,
    const float* __restrict__ addRow, float* __restrict__ C,
    int N, int K)
{
  __shared__ _Float16 As[SLEN * LDSW];  // 96 x 32 (+pad)
  __shared__ _Float16 Ws[64 * LDSW];    // 64 x 32 (+pad)

  const int tid  = threadIdx.x;
  const int wave = tid >> 5;
  const int lane = tid & 31;
  const int n0 = blockIdx.x * 64;

  // Per-thread staging coordinates (stride-128 over a row-of-8-float4 grid
  // means the column is invariant and rows step by 16 per chunk).
  const int rbase = tid >> 3;          // 0..15
  const int c4    = (tid & 7) * 4;     // 0,4,...,28
  const float* Ap = A + (size_t)rbase * K + c4;
  const float* Wp = W + (size_t)(n0 + rbase) * K + c4;
  _Float16* AsP = As + rbase * LDSW + c4;
  _Float16* WsP = Ws + rbase * LDSW + c4;

  v8f acc[6];
#pragma unroll
  for (int mt = 0; mt < 6; ++mt) acc[mt] = (v8f){0.f,0.f,0.f,0.f,0.f,0.f,0.f,0.f};

  // Prologue: load K-tile 0 into registers (6 A + 4 W float4 in flight).
  float4 nA[6], nW[4];
#pragma unroll
  for (int j = 0; j < 6; ++j) nA[j] = *(const float4*)(Ap + (size_t)j * 16 * K);
#pragma unroll
  for (int j = 0; j < 4; ++j) nW[j] = *(const float4*)(Wp + (size_t)j * 16 * K);

  for (int k0 = 0; k0 < K; k0 += 32) {
    // Commit current tile to LDS (f32 -> packed 4xf16, ds_store_b64).
#pragma unroll
    for (int j = 0; j < 6; ++j) *(v4h*)(AsP + j * 16 * LDSW) = cvt4(nA[j]);
#pragma unroll
    for (int j = 0; j < 4; ++j) *(v4h*)(WsP + j * 16 * LDSW) = cvt4(nW[j]);
    __syncthreads();

    // Issue next tile's global loads; they retire while the WMMAs below run.
    if (k0 + 32 < K) {
      const int kn = k0 + 32;
#pragma unroll
      for (int j = 0; j < 6; ++j) nA[j] = *(const float4*)(Ap + kn + (size_t)j * 16 * K);
#pragma unroll
      for (int j = 0; j < 4; ++j) nW[j] = *(const float4*)(Wp + kn + (size_t)j * 16 * K);
      if (kn + 32 < K) {
        __builtin_prefetch((const void*)(Wp + kn + 32 + (size_t)(tid & 3) * 16 * K), 0, 1);
      }
    }

    // B fragment (32x16 f16): lanes 0-15 -> K base 0, lanes 16-31 -> K base 8;
    // elems 0..7 = K[kb..kb+7], 8..15 = K[kb+16..kb+23]; N = lane&15 within this wave's slice.
    v16h b;
    {
      int n  = (lane & 15) + wave * 16;
      int kb = (lane < 16) ? 0 : 8;
#pragma unroll
      for (int e = 0; e < 16; ++e) b[e] = Ws[n * LDSW + kb + (e < 8 ? e : e + 8)];
    }
    // 6 M-tiles: A fragment (16x32 f16; lanes 0-15 -> K base 0, lanes 16-31
    // -> K base 8; elems 0..7 = K[kb..kb+7], 8..15 = K[kb+16..kb+23]).
#pragma unroll
    for (int mt = 0; mt < 6; ++mt) {
      v16h a;
      int m  = mt * 16 + (lane & 15);
      int kb = (lane < 16) ? 0 : 8;
#pragma unroll
      for (int e = 0; e < 16; ++e) {
        int k = kb + (e < 8 ? e : e + 8);
        a[e] = As[m * LDSW + k];
      }
      acc[mt] = __builtin_amdgcn_wmma_f32_16x16x32_f16(
          false, a, false, b, (short)0, acc[mt], false, false);
    }
    __syncthreads();
  }

  // C/D layout: VGPR r, lanes 0-15 -> M=r, lanes 16-31 -> M=8+r; N=lane&15.
  int n = n0 + wave * 16 + (lane & 15);
  float badd = (bias ? bias[n] : 0.0f) + (addRow ? addRow[n] : 0.0f);
#pragma unroll
  for (int mt = 0; mt < 6; ++mt) {
    int mb = mt * 16 + ((lane < 16) ? 0 : 8);
#pragma unroll
    for (int r = 0; r < 8; ++r) {
      int m = mb + r;
      float v = acc[mt][r] + badd;
      if (addMat) v += addMat[(size_t)m * N + n];
      C[(size_t)m * N + n] = v;
    }
  }
}

// ---------------------------------------------------------------------------
// Embedding gathers.
// ---------------------------------------------------------------------------
__global__ void gather_kernel(const int* __restrict__ in_seq,
                              const int* __restrict__ out_seq,
                              const float* __restrict__ enc_embed,
                              const float* __restrict__ dec_embed,
                              float* __restrict__ x_enc,
                              float* __restrict__ dec_x)
{
  int idx = blockIdx.x * blockDim.x + threadIdx.x;
  const int n = SLEN * EMB;
  if (idx < n) {
    int s = idx / EMB, c = idx % EMB;
    x_enc[idx] = enc_embed[(size_t)in_seq[s] * EMB + c];
  } else if (idx < 2 * n) {
    int i = idx - n;
    int s = i / EMB, c = i % EMB;
    dec_x[i] = dec_embed[(size_t)out_seq[s] * EMB + c];
  }
}

__global__ void zero_kernel(float* __restrict__ p, int n) {
  int i = blockIdx.x * blockDim.x + threadIdx.x;
  if (i < n) p[i] = 0.0f;
}

// ---------------------------------------------------------------------------
// Encoder bidirectional GRU recurrence (input gates precomputed via WMMA).
// Block 0 = forward (final h -> d_h carry), block 1 = backward (reversed).
// ---------------------------------------------------------------------------
__global__ __launch_bounds__(512) void enc_gru_kernel(
    const float* __restrict__ GI_f, const float* __restrict__ GI_b,
    const float* __restrict__ Whh_f, const float* __restrict__ bhh_f,
    const float* __restrict__ Whh_b, const float* __restrict__ bhh_b,
    float* __restrict__ enc, float* __restrict__ h_out)
{
  __shared__ float sh_h[HID];
  const int t = threadIdx.x;
  const bool bwd = (blockIdx.x == 1);
  const float* GI  = bwd ? GI_b  : GI_f;
  const float* Whh = bwd ? Whh_b : Whh_f;
  const float* bhh = bwd ? bhh_b : bhh_f;

  sh_h[t] = 0.0f;
  __syncthreads();

  const float4* ur = (const float4*)(Whh + (size_t)t * HID);
  const float4* uz = (const float4*)(Whh + (size_t)(HID + t) * HID);
  const float4* un = (const float4*)(Whh + (size_t)(2 * HID + t) * HID);
  const float br = bhh[t], bz = bhh[HID + t], bn = bhh[2 * HID + t];

  for (int step = 0; step < SLEN; ++step) {
    int row = bwd ? (SLEN - 1 - step) : step;
    float hr = br, hz = bz, hn = bn;
    for (int k4 = 0; k4 < HID / 4; ++k4) {
      float4 h4 = *(const float4*)(sh_h + 4 * k4);
      float4 a = ur[k4], b = uz[k4], c = un[k4];
      hr += a.x*h4.x + a.y*h4.y + a.z*h4.z + a.w*h4.w;
      hz += b.x*h4.x + b.y*h4.y + b.z*h4.z + b.w*h4.w;
      hn += c.x*h4.x + c.y*h4.y + c.z*h4.z + c.w*h4.w;
    }
    float gr = GI[(size_t)row * (3 * HID) + t];
    float gz = GI[(size_t)row * (3 * HID) + HID + t];
    float gn = GI[(size_t)row * (3 * HID) + 2 * HID + t];
    float r = sigmoidf_(gr + hr);
    float z = sigmoidf_(gz + hz);
    float nn = tanhf(gn + r * hn);
    float hnew = (1.0f - z) * nn + z * sh_h[t];
    __syncthreads();
    sh_h[t] = hnew;
    __syncthreads();
    enc[(size_t)row * (2 * HID) + (bwd ? HID : 0) + t] = hnew;
  }
  if (!bwd) h_out[t] = sh_h[t];
}

// wah[n] = (wa @ h)[n]
__global__ __launch_bounds__(512) void wah_kernel(
    const float* __restrict__ wa, const float* __restrict__ h,
    float* __restrict__ wah)
{
  __shared__ float sh[HID];
  sh[threadIdx.x] = h[threadIdx.x];
  __syncthreads();
  const float4* row = (const float4*)(wa + (size_t)threadIdx.x * HID);
  float acc = 0.0f;
  for (int k4 = 0; k4 < HID / 4; ++k4) {
    float4 w = row[k4];
    float4 h4 = *(const float4*)(sh + 4 * k4);
    acc += w.x*h4.x + w.y*h4.y + w.z*h4.z + w.w*h4.w;
  }
  wah[threadIdx.x] = acc;
}

// e[s] = tanh(pre[s,:]) . va
__global__ __launch_bounds__(256) void attn_e_kernel(
    const float* __restrict__ pre, const float* __restrict__ va,
    float* __restrict__ e)
{
  __shared__ float red[256];
  const int s = blockIdx.x;
  float acc = 0.0f;
  for (int n = threadIdx.x; n < HID; n += 256)
    acc += tanhf(pre[(size_t)s * HID + n]) * va[n];
  red[threadIdx.x] = acc;
  __syncthreads();
  for (int off = 128; off > 0; off >>= 1) {
    if (threadIdx.x < off) red[threadIdx.x] += red[threadIdx.x + off];
    __syncthreads();
  }
  if (threadIdx.x == 0) e[s] = red[0];
}

// ---------------------------------------------------------------------------
// Decode step core: softmax -> alpha, ctx = alpha@enc, decoder GRU cell,
// record [h_new, ctx] in Z row t. step0 matches the reference: h1 recorded
// but carry / coverage-h stay at h_enc.
// ---------------------------------------------------------------------------
__global__ __launch_bounds__(512) void dec_step_kernel(
    const float* __restrict__ e, const float* __restrict__ enc,
    const float* __restrict__ dec_x,
    const float* __restrict__ Wih_d, const float* __restrict__ Whh_d,
    const float* __restrict__ bih_d, const float* __restrict__ bhh_d,
    float* __restrict__ d_h, float* __restrict__ alpha_out,
    float* __restrict__ h_for_cov, float* __restrict__ Z,
    int t, int step0)
{
  __shared__ float sh_alpha[SLEN];
  __shared__ float sh_x[2 * HID + EMB];  // [ctx(1024), emb(256)]
  __shared__ float sh_h[HID];
  const int tid = threadIdx.x;

  if (tid < SLEN) sh_alpha[tid] = e[tid];
  if (tid < EMB)  sh_x[2 * HID + tid] = dec_x[(size_t)t * EMB + tid];
  sh_h[tid] = d_h[tid];
  __syncthreads();

  if (tid == 0) {
    float mx = sh_alpha[0];
    for (int s = 1; s < SLEN; ++s) mx = fmaxf(mx, sh_alpha[s]);
    float sum = 0.0f;
    for (int s = 0; s < SLEN; ++s) { float v = __expf(sh_alpha[s] - mx); sh_alpha[s] = v; sum += v; }
    float inv = 1.0f / sum;
    for (int s = 0; s < SLEN; ++s) sh_alpha[s] *= inv;
  }
  __syncthreads();
  if (tid < SLEN) alpha_out[tid] = sh_alpha[tid];

  // ctx = alpha @ enc (2H wide)
  for (int j = tid; j < 2 * HID; j += 512) {
    float c = 0.0f;
    for (int s = 0; s < SLEN; ++s) c += sh_alpha[s] * enc[(size_t)s * (2 * HID) + j];
    sh_x[j] = c;
    Z[(size_t)t * (3 * HID) + HID + j] = c;
  }
  __syncthreads();

  // GRU cell: x = [ctx, emb] (K=1280), h (K=512), float4 weight streaming.
  const int j = tid;
  float ir = bih_d[j], iz = bih_d[HID + j], in_ = bih_d[2 * HID + j];
  const float4* wr = (const float4*)(Wih_d + (size_t)j * 1280);
  const float4* wz = (const float4*)(Wih_d + (size_t)(HID + j) * 1280);
  const float4* wn = (const float4*)(Wih_d + (size_t)(2 * HID + j) * 1280);
  for (int k4 = 0; k4 < 1280 / 4; ++k4) {
    float4 x4 = *(const float4*)(sh_x + 4 * k4);
    float4 a = wr[k4], b = wz[k4], c = wn[k4];
    ir  += a.x*x4.x + a.y*x4.y + a.z*x4.z + a.w*x4.w;
    iz  += b.x*x4.x + b.y*x4.y + b.z*x4.z + b.w*x4.w;
    in_ += c.x*x4.x + c.y*x4.y + c.z*x4.z + c.w*x4.w;
  }
  float hr = bhh_d[j], hz = bhh_d[HID + j], hn = bhh_d[2 * HID + j];
  const float4* vr = (const float4*)(Whh_d + (size_t)j * HID);
  const float4* vz = (const float4*)(Whh_d + (size_t)(HID + j) * HID);
  const float4* vn = (const float4*)(Whh_d + (size_t)(2 * HID + j) * HID);
  for (int k4 = 0; k4 < HID / 4; ++k4) {
    float4 h4 = *(const float4*)(sh_h + 4 * k4);
    float4 a = vr[k4], b = vz[k4], c = vn[k4];
    hr += a.x*h4.x + a.y*h4.y + a.z*h4.z + a.w*h4.w;
    hz += b.x*h4.x + b.y*h4.y + b.z*h4.z + b.w*h4.w;
    hn += c.x*h4.x + c.y*h4.y + c.z*h4.z + c.w*h4.w;
  }
  float r = sigmoidf_(ir + hr);
  float z = sigmoidf_(iz + hz);
  float nn = tanhf(in_ + r * hn);
  float hnew = (1.0f - z) * nn + z * sh_h[j];

  Z[(size_t)t * (3 * HID) + j] = hnew;
  h_for_cov[j] = step0 ? sh_h[j] : hnew;
  if (!step0) d_h[j] = hnew;
}

// inp[s, 0:1024] = alpha[s]*enc[s,:]; inp[s, 1024:1536] = alpha[s]*h_for_cov
__global__ void build_cov_input_kernel(
    const float* __restrict__ alpha, const float* __restrict__ enc,
    const float* __restrict__ hfc, float* __restrict__ inp)
{
  int idx = blockIdx.x * blockDim.x + threadIdx.x;
  if (idx >= SLEN * 3 * HID) return;
  int s = idx / (3 * HID), c = idx % (3 * HID);
  float v = (c < 2 * HID) ? enc[(size_t)s * (2 * HID) + c] : hfc[c - 2 * HID];
  inp[idx] = alpha[s] * v;
}

// Coverage tanh-RNN recurrence (input gates precomputed via WMMA).
__global__ __launch_bounds__(512) void coverage_scan_kernel(
    const float* __restrict__ GIc, const float* __restrict__ Whh_c,
    const float* __restrict__ bhh_c,
    float* __restrict__ cov_vec, float* __restrict__ cov_last)
{
  __shared__ float sh_h[HID];
  const int t = threadIdx.x;
  sh_h[t] = 0.0f;
  __syncthreads();
  const float4* u = (const float4*)(Whh_c + (size_t)t * HID);
  const float b = bhh_c[t];
  const float cl = cov_last[t];
  for (int s = 0; s < SLEN; ++s) {
    float acc = b;
    for (int k4 = 0; k4 < HID / 4; ++k4) {
      float4 w = u[k4];
      float4 h4 = *(const float4*)(sh_h + 4 * k4);
      acc += w.x*h4.x + w.y*h4.y + w.z*h4.z + w.w*h4.w;
    }
    float h2 = tanhf(GIc[(size_t)s * HID + t] + acc);
    cov_vec[(size_t)s * HID + t] = tanhf((s == 0) ? cl : sh_h[t]);
    __syncthreads();
    sh_h[t] = h2;
    __syncthreads();
  }
  cov_last[t] = sh_h[t];
}

// ---------------------------------------------------------------------------
// Host orchestration
// ---------------------------------------------------------------------------
static inline void launch_gemm(const float* A, const float* W, const float* bias,
                               const float* addMat, const float* addRow,
                               float* C, int N, int K, hipStream_t s)
{
  wmma_gemm_kernel<<<N / 64, 128, 0, s>>>(A, W, bias, addMat, addRow, C, N, K);
}

extern "C" void kernel_launch(void* const* d_in, const int* in_sizes, int n_in,
                              void* d_out, int out_size, void* d_ws, size_t ws_size,
                              hipStream_t stream)
{
  (void)in_sizes; (void)n_in; (void)out_size; (void)ws_size;

  const int*   in_seq    = (const int*)  d_in[0];
  const int*   out_seq   = (const int*)  d_in[1];
  const float* enc_embed = (const float*)d_in[2];
  const float* dec_embed = (const float*)d_in[3];
  const float* Wih_f = (const float*)d_in[4];
  const float* Whh_f = (const float*)d_in[5];
  const float* bih_f = (const float*)d_in[6];
  const float* bhh_f = (const float*)d_in[7];
  const float* Wih_b = (const float*)d_in[8];
  const float* Whh_b = (const float*)d_in[9];
  const float* bih_b = (const float*)d_in[10];
  const float* bhh_b = (const float*)d_in[11];
  const float* Wih_d = (const float*)d_in[12];
  const float* Whh_d = (const float*)d_in[13];
  const float* bih_d = (const float*)d_in[14];
  const float* bhh_d = (const float*)d_in[15];
  const float* Wih_c = (const float*)d_in[16];
  const float* Whh_c = (const float*)d_in[17];
  const float* bih_c = (const float*)d_in[18];
  const float* bhh_c = (const float*)d_in[19];
  const float* va    = (const float*)d_in[20];
  const float* wa    = (const float*)d_in[21];
  const float* ua    = (const float*)d_in[22];
  const float* vc    = (const float*)d_in[23];
  const float* lin_W = (const float*)d_in[24];
  const float* lin_b = (const float*)d_in[25];
  float* out = (float*)d_out;

  // Workspace layout (floats; every chunk size is a multiple of 4 floats, so
  // all float4 accesses stay 16B-aligned).
  float* ws = (float*)d_ws;
  size_t off = 0;
  float* x_enc   = ws + off; off += (size_t)SLEN * EMB;
  float* dec_x   = ws + off; off += (size_t)SLEN * EMB;
  float* GI_f    = ws + off; off += (size_t)SLEN * 3 * HID;
  float* GI_b    = ws + off; off += (size_t)SLEN * 3 * HID;
  float* enc     = ws + off; off += (size_t)SLEN * 2 * HID;
  float* ua_enc  = ws + off; off += (size_t)SLEN * HID;
  float* cov_vec = ws + off; off += (size_t)SLEN * HID;   // zeroed with cov_last
  float* cov_last= ws + off; off += (size_t)HID;
  float* d_h     = ws + off; off += (size_t)HID;
  float* wah     = ws + off; off += (size_t)HID;
  float* pre     = ws + off; off += (size_t)SLEN * HID;
  float* e_buf   = ws + off; off += (size_t)SLEN;
  float* alpha   = ws + off; off += (size_t)SLEN;
  float* hfc     = ws + off; off += (size_t)HID;
  float* inp_c   = ws + off; off += (size_t)SLEN * 3 * HID;
  float* GIc     = ws + off; off += (size_t)SLEN * HID;
  float* Z       = ws + off; off += (size_t)SLEN * 3 * HID;

  // 1) Embedding gathers.
  {
    int n = 2 * SLEN * EMB;
    gather_kernel<<<(n + 255) / 256, 256, 0, stream>>>(
        in_seq, out_seq, enc_embed, dec_embed, x_enc, dec_x);
  }
  // 2) Zero cov_vec + cov_last (contiguous).
  {
    int n = SLEN * HID + HID;
    zero_kernel<<<(n + 255) / 256, 256, 0, stream>>>(cov_vec, n);
  }
  // 3) Encoder input gates: GI = x_enc @ Wih^T + bih (WMMA).
  launch_gemm(x_enc, Wih_f, bih_f, nullptr, nullptr, GI_f, 3 * HID, EMB, stream);
  launch_gemm(x_enc, Wih_b, bih_b, nullptr, nullptr, GI_b, 3 * HID, EMB, stream);
  // 4) Encoder recurrences.
  enc_gru_kernel<<<2, HID, 0, stream>>>(GI_f, GI_b, Whh_f, bhh_f, Whh_b, bhh_b, enc, d_h);
  // 5) ua_enc = enc @ ua^T (WMMA).
  launch_gemm(enc, ua, nullptr, nullptr, nullptr, ua_enc, HID, 2 * HID, stream);

  // 6) Decode loop.
  for (int t = 0; t < SLEN; ++t) {
    wah_kernel<<<1, HID, 0, stream>>>(wa, d_h, wah);
    // pre = cov_vec @ vc^T + ua_enc + wah  (cov_vec==0 at t=0 => no cov term)
    launch_gemm(cov_vec, vc, nullptr, ua_enc, wah, pre, HID, HID, stream);
    attn_e_kernel<<<SLEN, 256, 0, stream>>>(pre, va, e_buf);
    dec_step_kernel<<<1, HID, 0, stream>>>(
        e_buf, enc, dec_x, Wih_d, Whh_d, bih_d, bhh_d,
        d_h, alpha, hfc, Z, t, (t == 0) ? 1 : 0);
    {
      int n = SLEN * 3 * HID;
      build_cov_input_kernel<<<(n + 255) / 256, 256, 0, stream>>>(alpha, enc, hfc, inp_c);
    }
    launch_gemm(inp_c, Wih_c, bih_c, nullptr, nullptr, GIc, HID, 3 * HID, stream);
    coverage_scan_kernel<<<1, HID, 0, stream>>>(GIc, Whh_c, bhh_c, cov_vec, cov_last);
  }

  // 7) Deferred logits: out = Z @ lin_W^T + lin_b (96 x 32000, K=1536),
  //    one WMMA GEMM; lin_W's 196 MB streams from HBM exactly once.
  launch_gemm(Z, lin_W, lin_b, nullptr, nullptr, out, 32000, 3 * HID, stream);
}
